// non_local_denoising_block_17042430230629
// MI455X (gfx1250) — matrix operand order repaired and checked by reading, not compiled, not run on hardware
//
#include <hip/hip_runtime.h>

#define B_SZ   4
#define C_IN   256
#define C_I    128
#define N_PIX  4096
#define BN_EPS 1e-5f

typedef __attribute__((ext_vector_type(16))) __bf16 v16bf;
typedef __attribute__((ext_vector_type(8)))  __bf16 v8bf;
typedef __attribute__((ext_vector_type(8)))  float  v8f;

// A-fragment (16x32 bf16): lane holds row (lane%16); K = 2*kb..2*kb+15,
// kb = (lane<16) ? 0 : 8.  Two 16B chunks from a contiguous row.
__device__ __forceinline__ v16bf load_afrag(const __bf16* row, int kb) {
    v8bf lo = *(const v8bf*)(row + 2 * kb);
    v8bf hi = *(const v8bf*)(row + 2 * kb + 8);
    return __builtin_shufflevector(lo, hi, 0,1,2,3,4,5,6,7,8,9,10,11,12,13,14,15);
}

__device__ __forceinline__ v8f wmma_bf16(v16bf a, v16bf b, v8f c) {
    // (neg_a, A, neg_b, B, c_mod, C, reuse_a, reuse_b)
    return __builtin_amdgcn_wmma_f32_16x16x32_bf16(false, a, false, b, (short)0, c,
                                                   false, false);
}

// Flat LDS pointer -> wave-relative LDS byte offset (aperture: low 32 bits).
__device__ __forceinline__ unsigned lds_off32(const void* p) {
    return (unsigned)(uintptr_t)p;
}

// Async global->LDS copy of 64 bytes per lane (4 x B128 chunks, IOFFSET applies
// to both the LDS and global side).  Tracked by ASYNCcnt.
__device__ __forceinline__ void async_copy64(unsigned ldsAddr, const void* gAddr) {
    asm volatile(
        "global_load_async_to_lds_b128 %0, %1, off\n\t"
        "global_load_async_to_lds_b128 %0, %1, off offset:16\n\t"
        "global_load_async_to_lds_b128 %0, %1, off offset:32\n\t"
        "global_load_async_to_lds_b128 %0, %1, off offset:48"
        :: "v"(ldsAddr), "v"(gAddr) : "memory");
}

__device__ __forceinline__ void wait_async0() {
    asm volatile("s_wait_asynccnt 0x0" ::: "memory");
}

// ---------------------------------------------------------------- prep ------
__global__ void prep_kernel(const float* __restrict__ gw, const float* __restrict__ tw,
                            const float* __restrict__ pw, const float* __restrict__ zw,
                            __bf16* Wg, __bf16* Wt, __bf16* Wp, __bf16* Wz,
                            float* stats) {
    int i = blockIdx.x * 256 + threadIdx.x;          // CI*C = 32768 total
    if (i < C_I * C_IN) {
        Wg[i] = (__bf16)gw[i];
        Wt[i] = (__bf16)tw[i];
        Wp[i] = (__bf16)pw[i];
        Wz[i] = (__bf16)zw[i];
    }
    if (i < 2 * C_IN) stats[i] = 0.f;                // zero BN sum / sumsq
}

// ------------------------------------------------------------ transpose -----
// x [B,C,N] f32 -> xT [B,N,C] bf16 (GEMM-A friendly)
__global__ void xpose_kernel(const float* __restrict__ x, __bf16* __restrict__ xT) {
    size_t i = (size_t)blockIdx.x * 256 + threadIdx.x;      // B*C*N total
    size_t n = i % N_PIX;
    size_t c = (i / N_PIX) % C_IN;
    size_t b = i / ((size_t)N_PIX * C_IN);
    xT[(b * N_PIX + n) * C_IN + c] = (__bf16)x[i];
}

// ---------------------------------------------------- projection GEMMs ------
// out[n][ci] = sum_c xT[n][c] * W[ci][c] + bias[ci]
// layout 0: out [B,N,CI] row-major (theta, phi); layout 1: out [B,CI,N] (gT)
__global__ __launch_bounds__(128)
void proj_kernel(const __bf16* __restrict__ xT, const __bf16* __restrict__ W,
                 const float* __restrict__ bias, __bf16* __restrict__ out, int layout) {
    const int lane = threadIdx.x & 31;
    const int wave = threadIdx.x >> 5;
    const int l16  = lane & 15;
    const int hf   = lane >> 4;
    int gid = blockIdx.x * 4 + wave;                 // 8192 tiles
    int ciT = gid & 7;                               // CI/16
    int nT  = (gid >> 3) & 255;                      // N/16
    int b   = gid >> 11;

    const __bf16* arow = xT + ((size_t)b * N_PIX + nT * 16 + l16) * C_IN;
    int ci = ciT * 16 + l16;
    const __bf16* brow = W + (size_t)ci * C_IN + hf * 16;   // B col = W row (contig)

    v8f acc = {};
    #pragma unroll
    for (int kt = 0; kt < 8; ++kt) {                 // K = C_IN = 256
        v16bf a  = load_afrag(arow + kt * 32, hf * 8);
        v16bf bb = *(const v16bf*)(brow + kt * 32);
        acc = wmma_bf16(a, bb, acc);
    }
    float bv = bias[ci];
    #pragma unroll
    for (int v = 0; v < 8; ++v) {
        int r = v + hf * 8;
        float val = acc[v] + bv;
        if (layout == 0)
            out[((size_t)b * N_PIX + nT * 16 + r) * C_I + ci] = (__bf16)val;
        else
            out[((size_t)b * C_I + ci) * N_PIX + nT * 16 + r] = (__bf16)val;
    }
}

// --------------------------------------------- fused flash attention --------
// y[b,n,:] = softmax_m( theta[b,n,:] . phi[b,m,:] ) @ g[b,m,:]
__global__ __launch_bounds__(128)
void attn_kernel(const __bf16* __restrict__ theta, const __bf16* __restrict__ phi,
                 const __bf16* __restrict__ gT, __bf16* __restrict__ y) {
    __shared__ __align__(16) __bf16 sPhi[32 * C_I];     // [key 0..31][ci 0..127]
    __shared__ __align__(16) __bf16 sG[C_I * 32];       // [ci 0..127][key 0..31]
    __shared__ __align__(16) __bf16 sP[4][16 * 32];     // per-wave P tile

    const int tid  = threadIdx.x;
    const int wave = tid >> 5;
    const int lane = tid & 31;
    const int l16  = lane & 15;
    const int hf   = lane >> 4;
    const int kbA  = hf * 8;       // A-frag K base
    const int kbB  = hf * 16;      // B-frag K base

    const int rowBlk = blockIdx.x * 64;               // 64 query rows / block
    const int b  = rowBlk / N_PIX;
    const int n0 = (rowBlk % N_PIX) + wave * 16;

    // query fragments: 16 rows x 128 channels, kept in registers all loop
    v16bf aQ[4];
    {
        const __bf16* qrow = theta + ((size_t)b * N_PIX + n0 + l16) * C_I;
        #pragma unroll
        for (int kt = 0; kt < 4; ++kt) aQ[kt] = load_afrag(qrow + kt * 32, kbA);
    }

    float mrow[8], lrow[8];
    v8f oAcc[8];
    v8f zero = {};
    #pragma unroll
    for (int v = 0; v < 8; ++v) { mrow[v] = -1e30f; lrow[v] = 0.f; }
    #pragma unroll
    for (int j = 0; j < 8; ++j) oAcc[j] = zero;

    const __bf16* phiB = phi + (size_t)b * N_PIX * C_I;
    const __bf16* gB   = gT  + (size_t)b * C_I * N_PIX;

    // per-thread staging slots (constant across iterations)
    const int pr = tid >> 2, pcq = (tid & 3) * 32;    // phi: 32 rows x 128 ci
    const unsigned ldsPhiDst = lds_off32(sPhi + pr * C_I + pcq);
    const unsigned ldsGDst   = lds_off32(sG + tid * 32);   // g: 128 rows x 32 keys

    for (int mk = 0; mk < N_PIX; mk += 32) {
        // ---- async global->LDS staging of 32 keys of phi + g ---------------
        async_copy64(ldsPhiDst, phiB + (size_t)(mk + pr) * C_I + pcq);
        async_copy64(ldsGDst,   gB + (size_t)tid * N_PIX + mk);
        if (mk + 32 < N_PIX)
            __builtin_prefetch(phiB + (size_t)(mk + 32 + pr) * C_I, 0, 0);
        wait_async0();
        __syncthreads();

        // ---- S = theta_tile (16x128) @ phi_tile^T (128x32) -> two 16x16 ----
        v8f s0 = {}, s1 = {};
        #pragma unroll
        for (int kt = 0; kt < 4; ++kt) {
            v16bf b0 = *(const v16bf*)(sPhi + l16 * C_I + kt * 32 + kbB);
            v16bf b1 = *(const v16bf*)(sPhi + (16 + l16) * C_I + kt * 32 + kbB);
            s0 = wmma_bf16(aQ[kt], b0, s0);
            s1 = wmma_bf16(aQ[kt], b1, s1);
        }

        // ---- online softmax: row stats live in 16-lane half-groups ---------
        #pragma unroll
        for (int v = 0; v < 8; ++v) {
            float t = fmaxf(s0[v], s1[v]);
            t = fmaxf(t, __shfl_xor(t, 1, 32));
            t = fmaxf(t, __shfl_xor(t, 2, 32));
            t = fmaxf(t, __shfl_xor(t, 4, 32));
            t = fmaxf(t, __shfl_xor(t, 8, 32));
            float mn    = fmaxf(mrow[v], t);
            float alpha = __expf(mrow[v] - mn);
            mrow[v] = mn;
            float p0 = __expf(s0[v] - mn);
            float p1 = __expf(s1[v] - mn);
            float ps = p0 + p1;
            ps += __shfl_xor(ps, 1, 32);
            ps += __shfl_xor(ps, 2, 32);
            ps += __shfl_xor(ps, 4, 32);
            ps += __shfl_xor(ps, 8, 32);
            lrow[v] = lrow[v] * alpha + ps;
            #pragma unroll
            for (int j = 0; j < 8; ++j) oAcc[j][v] *= alpha;
            int r = v + hf * 8;                        // accumulator-layout row
            sP[wave][r * 32 + l16]      = (__bf16)p0;  // P -> LDS (A-layout reload)
            sP[wave][r * 32 + 16 + l16] = (__bf16)p1;
        }
        __syncthreads();

        // ---- O += P (16x32) @ g_tile (32x128) ------------------------------
        v16bf pf = load_afrag(&sP[wave][l16 * 32], kbA);
        #pragma unroll
        for (int j = 0; j < 8; ++j) {
            v16bf bg = *(const v16bf*)(sG + (j * 16 + l16) * 32 + kbB);
            oAcc[j] = wmma_bf16(pf, bg, oAcc[j]);
        }
        __syncthreads();
    }

    // ---- epilogue: y = O / l, store [B,N,CI] bf16 --------------------------
    #pragma unroll
    for (int j = 0; j < 8; ++j) {
        #pragma unroll
        for (int v = 0; v < 8; ++v) {
            int r = v + hf * 8;
            float val = oAcc[j][v] / lrow[v];
            y[((size_t)b * N_PIX + n0 + r) * C_I + j * 16 + l16] = (__bf16)val;
        }
    }
}

// ------------------------------------------- output projection + BN stats ---
__global__ __launch_bounds__(128)
void outproj_kernel(const __bf16* __restrict__ y, const __bf16* __restrict__ Wz,
                    const float* __restrict__ bias, float* __restrict__ out,
                    float* __restrict__ stats) {
    const int lane = threadIdx.x & 31;
    const int wave = threadIdx.x >> 5;
    const int l16  = lane & 15;
    const int hf   = lane >> 4;
    int gid = blockIdx.x * 4 + wave;                 // 16384 tiles
    int oT = gid & 15;                               // C/16
    int nT = (gid >> 4) & 255;                       // N/16
    int b  = gid >> 12;

    const __bf16* arow = y + ((size_t)b * N_PIX + nT * 16 + l16) * C_I;
    int o = oT * 16 + l16;
    const __bf16* brow = Wz + (size_t)o * C_I + hf * 16;

    v8f acc = {};
    #pragma unroll
    for (int kt = 0; kt < 4; ++kt) {                 // K = CI = 128
        v16bf a  = load_afrag(arow + kt * 32, hf * 8);
        v16bf bb = *(const v16bf*)(brow + kt * 32);
        acc = wmma_bf16(a, bb, acc);
    }
    float bv = bias[o];
    float s1 = 0.f, s2 = 0.f;
    #pragma unroll
    for (int v = 0; v < 8; ++v) {
        int r = v + hf * 8;
        float val = acc[v] + bv;
        out[((size_t)b * C_IN + o) * N_PIX + nT * 16 + r] = val;   // w_y pre-BN
        s1 += val;
        s2 += val * val;
    }
    s1 += __shfl_xor(s1, 16, 32);                    // merge both half-groups (same col)
    s2 += __shfl_xor(s2, 16, 32);
    if (hf == 0) {
        atomicAdd(&stats[o], s1);
        atomicAdd(&stats[C_IN + o], s2);
    }
}

// ----------------------------------------------- BN finalize + residual -----
__global__ void bnres_kernel(const float* __restrict__ x, const float* __restrict__ gamma,
                             const float* __restrict__ beta, float* __restrict__ out,
                             const float* __restrict__ stats) {
    size_t i = (size_t)blockIdx.x * 256 + threadIdx.x;      // B*C*N total
    int c = (int)((i / N_PIX) % C_IN);
    const float cnt = (float)(B_SZ * N_PIX);
    float mean = stats[c] / cnt;
    float var  = stats[C_IN + c] / cnt - mean * mean;       // biased var
    float sc   = gamma[c] * rsqrtf(var + BN_EPS);
    out[i] = sc * (out[i] - mean) + beta[c] + x[i];
}

// ---------------------------------------------------------------------------
extern "C" void kernel_launch(void* const* d_in, const int* in_sizes, int n_in,
                              void* d_out, int out_size, void* d_ws, size_t ws_size,
                              hipStream_t stream) {
    const float* x       = (const float*)d_in[0];
    const float* g_w     = (const float*)d_in[1];
    const float* g_b     = (const float*)d_in[2];
    const float* theta_w = (const float*)d_in[3];
    const float* theta_b = (const float*)d_in[4];
    const float* phi_w   = (const float*)d_in[5];
    const float* phi_b   = (const float*)d_in[6];
    const float* wz_w    = (const float*)d_in[7];
    const float* wz_b    = (const float*)d_in[8];
    const float* bn_g    = (const float*)d_in[9];
    const float* bn_b    = (const float*)d_in[10];
    float* out = (float*)d_out;

    // workspace carve-up (256B aligned slices)
    char*  ws  = (char*)d_ws;
    size_t off = 0;
    auto take = [&](size_t bytes) { size_t o = off; off += (bytes + 255) & ~(size_t)255; return o; };
    __bf16* xT  = (__bf16*)(ws + take((size_t)B_SZ * N_PIX * C_IN * 2));   // 8 MB
    __bf16* th  = (__bf16*)(ws + take((size_t)B_SZ * N_PIX * C_I * 2));    // 4 MB
    __bf16* ph  = (__bf16*)(ws + take((size_t)B_SZ * N_PIX * C_I * 2));    // 4 MB
    __bf16* gTb = (__bf16*)(ws + take((size_t)B_SZ * C_I * N_PIX * 2));    // 4 MB
    __bf16* yv  = (__bf16*)(ws + take((size_t)B_SZ * N_PIX * C_I * 2));    // 4 MB
    __bf16* Wg  = (__bf16*)(ws + take((size_t)C_I * C_IN * 2));
    __bf16* Wt  = (__bf16*)(ws + take((size_t)C_I * C_IN * 2));
    __bf16* Wp  = (__bf16*)(ws + take((size_t)C_I * C_IN * 2));
    __bf16* Wz  = (__bf16*)(ws + take((size_t)C_IN * C_I * 2));
    float*  st  = (float*)(ws + take((size_t)2 * C_IN * 4));

    // 1) weight conversion + stats zeroing (deterministic per call)
    prep_kernel<<<128, 256, 0, stream>>>(g_w, theta_w, phi_w, wz_w, Wg, Wt, Wp, Wz, st);
    // 2) x -> xT bf16
    xpose_kernel<<<(B_SZ * C_IN * N_PIX) / 256, 256, 0, stream>>>(x, xT);
    // 3) projections (WMMA GEMMs)
    proj_kernel<<<2048, 128, 0, stream>>>(xT, Wt, theta_b, th, 0);
    proj_kernel<<<2048, 128, 0, stream>>>(xT, Wp, phi_b, ph, 0);
    proj_kernel<<<2048, 128, 0, stream>>>(xT, Wg, g_b, gTb, 1);
    // 4) fused flash attention (async global->LDS staging)
    attn_kernel<<<(B_SZ * N_PIX) / 64, 128, 0, stream>>>(th, ph, gTb, yv);
    // 5) output projection + BN statistics
    outproj_kernel<<<4096, 128, 0, stream>>>(yv, Wz, wz_b, out, st);
    // 6) BN + residual
    bnres_kernel<<<(B_SZ * C_IN * N_PIX) / 256, 256, 0, stream>>>(x, bn_g, bn_b, out, st);
}
